// LSTM_VAE_8589934592459
// MI455X (gfx1250) — matrix operand order repaired and checked by reading, not compiled, not run on hardware
//
#include <hip/hip_runtime.h>
#include <hip/hip_bf16.h>
#include <math.h>

typedef _Float16 f16;
typedef __attribute__((ext_vector_type(16))) _Float16 v16h;
typedef __attribute__((ext_vector_type(8)))  float    v8f;
typedef __attribute__((ext_vector_type(4)))  unsigned u32x4;

#define GAS __attribute__((address_space(1)))   // global
#define LAS __attribute__((address_space(3)))   // LDS

#define HIDN 256
#define HSTR 268   // padded LDS row stride (halves): 134 dwords -> conflict-free A loads
#define NB   1024
#define TSEQ 65
#define BLK  512   // 16 waves of 32

union AFrag { v16h v; unsigned u[8]; };

__device__ __forceinline__ float sigm(float x) { return 1.f / (1.f + __expf(-x)); }

// Defeat LICM so weight-tile loads are not hoisted out of the timestep loop
// (hoisting caused scratch spills). Address space is re-annotated at load sites.
__device__ __forceinline__ const f16* opaque(const f16* p) {
  unsigned long long q = (unsigned long long)p;
  asm volatile("" : "+v"(q));
  return (const f16*)q;
}

// A-fragment (16x32 f16) from row-major GLOBAL buffer, base -> element (m=0,k=0).
// A layout (ISA 7.12.2): lanes 0-15 rows M=0..15; per lane halves e=0..15:
//   k = (e%8) + 16*(e/8) + 8*(lane/16)
__device__ __forceinline__ v16h load_A_g(const f16* base, int rowstride, int lane) {
  AFrag a;
  const int m  = lane & 15;
  const int hi = lane >> 4;
  const GAS unsigned* p = (const GAS unsigned*)(base + m * rowstride + 8 * hi);
#pragma unroll
  for (int v = 0; v < 4; ++v) a.u[v] = p[v];       // halves 2v,2v+1   -> k=2v,2v+1
#pragma unroll
  for (int v = 4; v < 8; ++v) a.u[v] = p[v + 4];   // halves 8+2v,...  -> k=2v+8,2v+9
  return a.v;
}

// Same, from LDS.
__device__ __forceinline__ v16h load_A_l(const f16* base, int rowstride, int lane) {
  AFrag a;
  const int m  = lane & 15;
  const int hi = lane >> 4;
  const LAS unsigned* p = (const LAS unsigned*)(base + m * rowstride + 8 * hi);
#pragma unroll
  for (int v = 0; v < 4; ++v) a.u[v] = p[v];
#pragma unroll
  for (int v = 4; v < 8; ++v) a.u[v] = p[v + 4];
  return a.v;
}

// Pre-packed 32x16 B-fragment from GLOBAL: packed[((kt*64+nt)*32+lane)*16 + e]
__device__ __forceinline__ v16h load_B_pk(const f16* pk, int kt, int nt, int lane) {
  const GAS u32x4* q = (const GAS u32x4*)(pk + ((((size_t)kt * 64 + nt) * 32 + lane) << 4));
  u32x4 x0 = q[0], x1 = q[1];
  AFrag a;
  a.u[0] = x0.x; a.u[1] = x0.y; a.u[2] = x0.z; a.u[3] = x0.w;
  a.u[4] = x1.x; a.u[5] = x1.y; a.u[6] = x1.z; a.u[7] = x1.w;
  return a.v;
}

#define WMMA(A, B, C) __builtin_amdgcn_wmma_f32_16x16x32_f16(false, (A), false, (B), (short)0, (C), false, false)

// Pack f32 weight matrix W[1024 rows][rowstride cols] into B-fragment order:
// B[k][n] = W[n][k], zero-padded past kreal.
__global__ void pack_b_kernel(const float* __restrict__ src, f16* __restrict__ dst,
                              int rowstride, int kreal, int total) {
  int idx = blockIdx.x * 256 + threadIdx.x;
  if (idx >= total) return;
  int e    = idx & 15;
  int lane = (idx >> 4) & 31;
  int nt   = (idx >> 9) & 63;
  int kt   = idx >> 15;
  int n = nt * 16 + (lane & 15);
  int k = kt * 32 + (e & 7) + 16 * (e >> 3) + 8 * (lane >> 4);
  float v = (k < kreal) ? src[(size_t)n * rowstride + k] : 0.f;
  dst[idx] = (f16)v;
}

// X [B][7][65] f32 -> x0pad [T][B][32] f16 (cols 7..31 zero)
__global__ void convert_x_kernel(const float* __restrict__ X, f16* __restrict__ xp) {
  int idx = blockIdx.x * 256 + threadIdx.x;
  if (idx >= TSEQ * NB * 32) return;
  int c = idx & 31;
  int b = (idx >> 5) & 1023;
  int t = idx >> 15;
  xp[idx] = (f16)((c < 7) ? X[((size_t)b * 7 + c) * TSEQ + t] : 0.f);
}

__global__ void vadd_kernel(const float* __restrict__ a, const float* __restrict__ b,
                            float* __restrict__ o, int n) {
  int i = blockIdx.x * 256 + threadIdx.x;
  if (i < n) o[i] = a[i] + b[i];
}

// One encoder LSTM layer, both directions (blockIdx.y = dir; dir1 runs reversed).
// Block = 32 batch rows, 16 waves; wave (rg,wv) owns row-group rg (16 rows) and
// 32 columns of each gate block -> 8 output tiles = 64 acc VGPRs per wave.
__global__ __launch_bounds__(BLK) void lstm_enc_kernel(
    const f16* __restrict__ x_seq, int XK, int wih_stride,
    const f16* __restrict__ wih_pk, const f16* __restrict__ whh_pk,
    const float* __restrict__ bias2,      // [2][1024]  (bih+bhh)
    f16* __restrict__ y_seq,              // [T][B][512], write at col dir*256
    float* __restrict__ summary, int slot_base) {
  const int tid  = threadIdx.x;
  const int wave = tid >> 5, lane = tid & 31;
  const int wv = wave & 7, rg = wave >> 3;
  const int dir  = blockIdx.y;
  const int b0   = blockIdx.x * 32;
  const f16* wih0 = wih_pk + (size_t)dir * wih_stride;
  const f16* whh0 = whh_pk + (size_t)dir * 262144;

  __shared__ f16   h_buf[32 * HSTR];
  __shared__ float c_buf[32 * HIDN];
  __shared__ float bias_s[1024];
  for (int i = tid; i < 32 * HSTR; i += BLK) h_buf[i] = (f16)0.f;
  for (int i = tid; i < 32 * HIDN; i += BLK) c_buf[i] = 0.f;
  for (int i = tid; i < 1024; i += BLK) bias_s[i] = bias2[dir * 1024 + i];
  __syncthreads();

  const int KTX = XK >> 5;
  const int hi = lane >> 4, ln = lane & 15;

  for (int step = 0; step < TSEQ; ++step) {
    const int t = dir ? (TSEQ - 1 - step) : step;
    const f16* wih = opaque(wih0);
    const f16* whh = opaque(whh0);
    v8f acc[4][2];
    v8f zero = {};
#pragma unroll
    for (int g = 0; g < 4; ++g)
#pragma unroll
      for (int s = 0; s < 2; ++s) acc[g][s] = zero;

    const f16* xb = x_seq + ((size_t)t * NB + b0 + rg * 16) * XK;
    for (int kt = 0; kt < KTX; ++kt) {          // input part
      v16h a = load_A_g(xb + kt * 32, XK, lane);
#pragma unroll
      for (int g = 0; g < 4; ++g) {
        v16h bA = load_B_pk(wih, kt, g * 16 + wv * 2 + 0, lane);
        v16h bB = load_B_pk(wih, kt, g * 16 + wv * 2 + 1, lane);
        acc[g][0] = WMMA(a, bA, acc[g][0]);
        acc[g][1] = WMMA(a, bB, acc[g][1]);
      }
    }
#pragma unroll
    for (int kt = 0; kt < 8; ++kt) {            // recurrent part
      v16h a = load_A_l(h_buf + rg * 16 * HSTR + kt * 32, HSTR, lane);
#pragma unroll
      for (int g = 0; g < 4; ++g) {
        v16h bA = load_B_pk(whh, kt, g * 16 + wv * 2 + 0, lane);
        v16h bB = load_B_pk(whh, kt, g * 16 + wv * 2 + 1, lane);
        acc[g][0] = WMMA(a, bA, acc[g][0]);
        acc[g][1] = WMMA(a, bB, acc[g][1]);
      }
    }
    __syncthreads();   // all reads of h_buf done before overwrite
#pragma unroll
    for (int s = 0; s < 2; ++s) {
      const int j = wv * 32 + s * 16 + ln;
#pragma unroll
      for (int r = 0; r < 8; ++r) {
        const int m = r + 8 * hi + rg * 16;
        float iv = acc[0][s][r] + bias_s[j];
        float fv = acc[1][s][r] + bias_s[256 + j];
        float gv = acc[2][s][r] + bias_s[512 + j];
        float ov = acc[3][s][r] + bias_s[768 + j];
        float c = c_buf[m * HIDN + j];
        c = sigm(fv) * c + sigm(iv) * tanhf(gv);
        c_buf[m * HIDN + j] = c;
        float h = sigm(ov) * tanhf(c);
        h_buf[m * HSTR + j] = (f16)h;
      }
    }
    __syncthreads();
    // coalesced hidden-state writeback: 32 bytes per thread from LDS
    {
      const int m = tid >> 4;            // 512 threads * 16 halves = 8192 = 32*256
      const int j0 = (tid & 15) * 16;
      const uint2* s2 = (const uint2*)(h_buf + m * HSTR + j0);
      uint2* d2 = (uint2*)(y_seq + ((size_t)t * NB + b0 + m) * 512 + dir * 256 + j0);
      d2[0] = s2[0]; d2[1] = s2[1]; d2[2] = s2[2]; d2[3] = s2[3];
    }
  }
  const int slot = slot_base + dir;
  for (int i = tid; i < 32 * HIDN; i += BLK) {
    int m = i >> 8, j = i & 255;
    summary[(size_t)(b0 + m) * 1536 + slot * 256 + j] = (float)h_buf[m * HSTR + j];
  }
}

__global__ void fcmu_kernel(const float* __restrict__ summary,
                            const float* __restrict__ Wmu, const float* __restrict__ bmu,
                            const float* __restrict__ Wlv, const float* __restrict__ blv,
                            const float* __restrict__ eps,
                            float* __restrict__ mu_out, float* __restrict__ lv_out,
                            float* __restrict__ z) {
  int idx = blockIdx.x * 256 + threadIdx.x;
  if (idx >= NB * 128) return;
  int b = idx >> 7, o = idx & 127;
  const float* srow = summary + (size_t)b * 1536;
  const float* wm = Wmu + (size_t)o * 1536;
  const float* wl = Wlv + (size_t)o * 1536;
  float sm = bmu[o], sl = blv[o];
  for (int k = 0; k < 1536; ++k) { float x = srow[k]; sm += x * wm[k]; sl += x * wl[k]; }
  mu_out[idx] = sm;
  lv_out[idx] = sl;
  z[idx] = sm + eps[idx] * __expf(0.5f * sl);
}

__global__ void mvp_kernel(const float* __restrict__ z,
                           const float* __restrict__ W1, const float* __restrict__ b1,
                           const float* __restrict__ W2, const float* __restrict__ b2,
                           const float* __restrict__ W3, const float* __restrict__ b3,
                           float* __restrict__ maxv) {
  int b = blockIdx.x * 64 + threadIdx.x;
  if (b >= NB) return;
  const float* zr = z + (size_t)b * 128;
  float m1[30], m2[30];
  for (int i = 0; i < 30; ++i) {
    float s = b1[i];
    for (int k = 0; k < 128; ++k) s += zr[k] * W1[i * 128 + k];
    m1[i] = s * sigm(s);
  }
  for (int i = 0; i < 30; ++i) {
    float s = b2[i];
    for (int k = 0; k < 30; ++k) s += m1[k] * W2[i * 30 + k];
    m2[i] = s * sigm(s);
  }
  for (int c = 0; c < 7; ++c) {
    float s = b3[c];
    for (int k = 0; k < 30; ++k) s += m2[k] * W3[c * 30 + k];
    maxv[b * 7 + c] = s;
  }
}

// M = z@W^T + b for h0 and c0. The torch .view(3,B,256) of the [B,768] result is
// a raw flat reinterpret, so the output buffer IS [3][B][256].
__global__ void l2hc_kernel(const float* __restrict__ z,
                            const float* __restrict__ Wh, const float* __restrict__ bh,
                            const float* __restrict__ Wc, const float* __restrict__ bc,
                            float* __restrict__ h0, float* __restrict__ c0) {
  int idx = blockIdx.x * 256 + threadIdx.x;
  if (idx >= NB * 768) return;
  int b = idx / 768, n = idx % 768;
  const float* zr = z + (size_t)b * 128;
  const float* wh = Wh + (size_t)n * 128;
  const float* wc = Wc + (size_t)n * 128;
  float sh = bh[n], sc = bc[n];
  for (int k = 0; k < 128; ++k) { float x = zr[k]; sh += x * wh[k]; sc += x * wc[k]; }
  h0[idx] = sh;
  c0[idx] = sc;
}

// zih[b][n] = z[b] . dec_Wih0[n][7:135]  (constant per-step contribution of z)
__global__ void zih_kernel(const float* __restrict__ z, const float* __restrict__ Wih0,
                           float* __restrict__ zih) {
  int idx = blockIdx.x * 256 + threadIdx.x;
  if (idx >= NB * 1024) return;
  int b = idx >> 10, n = idx & 1023;
  const float* zr = z + (size_t)b * 128;
  const float* w = Wih0 + (size_t)n * 135 + 7;
  float s = 0.f;
  for (int k = 0; k < 128; ++k) s += zr[k] * w[k];
  zih[idx] = s;
}

// Autoregressive 3-layer decoder; one block per 32 batch rows, 65 steps, 16 waves.
#define X0STR 34
__global__ __launch_bounds__(BLK) void lstm_dec_kernel(
    const f16* __restrict__ wp_pk,         // layer0 prev-out weights (K=7 pad 32)
    const f16* __restrict__ wih12_pk,      // [2][8][64] packed
    const f16* __restrict__ whh_pk,        // [3][8][64] packed
    const float* __restrict__ bias3,       // [3][1024]
    const float* __restrict__ zih,         // [B][1024]
    const float* __restrict__ h0s, const float* __restrict__ c0s,  // [3][B][256]
    const float* __restrict__ outW, const float* __restrict__ outb,
    float* __restrict__ Xhat) {            // [B][7][65]
  const int tid  = threadIdx.x;
  const int wave = tid >> 5, lane = tid & 31;
  const int wv = wave & 7, rg = wave >> 3;
  const int b0   = blockIdx.x * 32;
  const int hi = lane >> 4, ln = lane & 15;

  __shared__ f16   h_buf[3][32 * HSTR];
  __shared__ float c_buf[3][32 * HIDN];
  __shared__ f16   x0_buf[32 * X0STR];
  __shared__ float bias_s[3 * 1024];

  for (int i = tid; i < 3 * 32 * HIDN; i += BLK) {
    int l = i / (32 * HIDN), rem = i % (32 * HIDN);
    int m = rem >> 8, j = rem & 255;
    h_buf[l][m * HSTR + j] = (f16)h0s[((size_t)l * NB + b0 + m) * HIDN + j];
    c_buf[l][rem] = c0s[((size_t)l * NB + b0 + m) * HIDN + j];
  }
  for (int i = tid; i < 32 * X0STR; i += BLK) x0_buf[i] = (f16)0.f;
  for (int i = tid; i < 3 * 1024; i += BLK) bias_s[i] = bias3[i];
  __syncthreads();

  for (int t = 0; t < TSEQ; ++t) {
    for (int l = 0; l < 3; ++l) {
      v8f acc[4][2];
      v8f zero = {};
#pragma unroll
      for (int g = 0; g < 4; ++g)
#pragma unroll
        for (int s = 0; s < 2; ++s) acc[g][s] = zero;

      if (l == 0) {          // prev-output part (K=32)
        const f16* wp = opaque(wp_pk);
        v16h a = load_A_l(x0_buf + rg * 16 * X0STR, X0STR, lane);
#pragma unroll
        for (int g = 0; g < 4; ++g) {
          v16h bA = load_B_pk(wp, 0, g * 16 + wv * 2 + 0, lane);
          v16h bB = load_B_pk(wp, 0, g * 16 + wv * 2 + 1, lane);
          acc[g][0] = WMMA(a, bA, acc[g][0]);
          acc[g][1] = WMMA(a, bB, acc[g][1]);
        }
      } else {               // lower-layer hidden of current step (K=256)
        const f16* wih = opaque(wih12_pk + (size_t)(l - 1) * 262144);
#pragma unroll
        for (int kt = 0; kt < 8; ++kt) {
          v16h a = load_A_l(&h_buf[l - 1][rg * 16 * HSTR + kt * 32], HSTR, lane);
#pragma unroll
          for (int g = 0; g < 4; ++g) {
            v16h bA = load_B_pk(wih, kt, g * 16 + wv * 2 + 0, lane);
            v16h bB = load_B_pk(wih, kt, g * 16 + wv * 2 + 1, lane);
            acc[g][0] = WMMA(a, bA, acc[g][0]);
            acc[g][1] = WMMA(a, bB, acc[g][1]);
          }
        }
      }
      const f16* whh = opaque(whh_pk + (size_t)l * 262144);   // recurrent part
#pragma unroll
      for (int kt = 0; kt < 8; ++kt) {
        v16h a = load_A_l(&h_buf[l][rg * 16 * HSTR + kt * 32], HSTR, lane);
#pragma unroll
        for (int g = 0; g < 4; ++g) {
          v16h bA = load_B_pk(whh, kt, g * 16 + wv * 2 + 0, lane);
          v16h bB = load_B_pk(whh, kt, g * 16 + wv * 2 + 1, lane);
          acc[g][0] = WMMA(a, bA, acc[g][0]);
          acc[g][1] = WMMA(a, bB, acc[g][1]);
        }
      }
      __syncthreads();
#pragma unroll
      for (int s = 0; s < 2; ++s) {
        const int j = wv * 32 + s * 16 + ln;
#pragma unroll
        for (int r = 0; r < 8; ++r) {
          const int m = r + 8 * hi + rg * 16;
          float zi = 0.f, zf = 0.f, zg = 0.f, zo = 0.f;
          if (l == 0) {
            const float* zr = zih + (size_t)(b0 + m) * 1024;
            zi = zr[j]; zf = zr[256 + j]; zg = zr[512 + j]; zo = zr[768 + j];
          }
          float iv = acc[0][s][r] + bias_s[l * 1024 + j] + zi;
          float fv = acc[1][s][r] + bias_s[l * 1024 + 256 + j] + zf;
          float gv = acc[2][s][r] + bias_s[l * 1024 + 512 + j] + zg;
          float ov = acc[3][s][r] + bias_s[l * 1024 + 768 + j] + zo;
          float c = c_buf[l][m * HIDN + j];
          c = sigm(fv) * c + sigm(iv) * tanhf(gv);
          c_buf[l][m * HIDN + j] = c;
          float h = sigm(ov) * tanhf(c);
          h_buf[l][m * HSTR + j] = (f16)h;
        }
      }
      __syncthreads();
    }
    // output projection: raw value feeds back, clipped value is stored
    if (tid < 224) {
      int m = tid / 7, cc = tid % 7;
      float s = outb[cc];
      const f16* hr = &h_buf[2][m * HSTR];
      const float* w = outW + cc * HIDN;
      for (int k = 0; k < HIDN; ++k) s += (float)hr[k] * w[k];
      x0_buf[m * X0STR + cc] = (f16)s;
      Xhat[((size_t)(b0 + m) * 7 + cc) * TSEQ + t] = fminf(fmaxf(s, 0.f), 1.f);
    }
    __syncthreads();
  }
}

extern "C" void kernel_launch(void* const* d_in, const int* in_sizes, int n_in,
                              void* d_out, int out_size, void* d_ws, size_t ws_size,
                              hipStream_t stream) {
  const float* X        = (const float*)d_in[0];
  const float* eps      = (const float*)d_in[1];
  const float* encWih0  = (const float*)d_in[2];
  const float* encWhh0  = (const float*)d_in[3];
  const float* encBih0  = (const float*)d_in[4];
  const float* encBhh0  = (const float*)d_in[5];
  const float* encWih12 = (const float*)d_in[6];
  const float* encWhh12 = (const float*)d_in[7];
  const float* encBih12 = (const float*)d_in[8];
  const float* encBhh12 = (const float*)d_in[9];
  const float* fcMuW = (const float*)d_in[10];
  const float* fcMuB = (const float*)d_in[11];
  const float* fcLvW = (const float*)d_in[12];
  const float* fcLvB = (const float*)d_in[13];
  const float* mvpW1 = (const float*)d_in[14];
  const float* mvpB1 = (const float*)d_in[15];
  const float* mvpW2 = (const float*)d_in[16];
  const float* mvpB2 = (const float*)d_in[17];
  const float* mvpW3 = (const float*)d_in[18];
  const float* mvpB3 = (const float*)d_in[19];
  const float* l2hW = (const float*)d_in[20];
  const float* l2hB = (const float*)d_in[21];
  const float* l2cW = (const float*)d_in[22];
  const float* l2cB = (const float*)d_in[23];
  const float* decWih0  = (const float*)d_in[24];
  const float* decWih12 = (const float*)d_in[25];
  const float* decWhh   = (const float*)d_in[26];
  const float* decBih   = (const float*)d_in[27];
  const float* decBhh   = (const float*)d_in[28];
  const float* outW = (const float*)d_in[29];
  const float* outB = (const float*)d_in[30];

  float* Xhat = (float*)d_out;            // [1024][7][65] = 465920
  float* muO  = (float*)d_out + 465920;   // [1024][128]
  float* lvO  = muO + 131072;
  float* mxO  = lvO + 131072;             // [1024][7]

  char* wsb = (char*)d_ws;
  size_t off = 0;
  auto alloc = [&](size_t bytes) -> void* {
    off = (off + 255) & ~(size_t)255;
    void* p = wsb + off;
    off += bytes;
    return p;
  };
  f16*   x0pad  = (f16*)alloc((size_t)TSEQ * NB * 32 * 2);
  f16*   seqA   = (f16*)alloc((size_t)TSEQ * NB * 512 * 2);
  f16*   seqB   = (f16*)alloc((size_t)TSEQ * NB * 512 * 2);
  f16*   eW0pk  = (f16*)alloc((size_t)2 * 32768 * 2);
  f16*   eH0pk  = (f16*)alloc((size_t)2 * 262144 * 2);
  f16*   eW12pk = (f16*)alloc((size_t)4 * 524288 * 2);
  f16*   eH12pk = (f16*)alloc((size_t)4 * 262144 * 2);
  f16*   dWPpk  = (f16*)alloc((size_t)32768 * 2);
  f16*   dW12pk = (f16*)alloc((size_t)2 * 262144 * 2);
  f16*   dHpk   = (f16*)alloc((size_t)3 * 262144 * 2);
  float* eBias  = (float*)alloc((size_t)6144 * 4);   // [3][2][1024]
  float* dBias  = (float*)alloc((size_t)3072 * 4);   // [3][1024]
  float* summ   = (float*)alloc((size_t)NB * 1536 * 4);
  float* zbuf   = (float*)alloc((size_t)NB * 128 * 4);
  float* h0s    = (float*)alloc((size_t)786432 * 4); // == [3][B][256] raw view
  float* c0s    = (float*)alloc((size_t)786432 * 4);
  float* zihb   = (float*)alloc((size_t)NB * 1024 * 4);

  auto cdiv = [](int a, int b) { return (a + b - 1) / b; };

  // --- weight packing + input conversion + bias combine ---
  for (int d = 0; d < 2; ++d) {
    pack_b_kernel<<<cdiv(32768, 256), 256, 0, stream>>>(encWih0 + (size_t)d * 1024 * 7, eW0pk + (size_t)d * 32768, 7, 7, 32768);
    pack_b_kernel<<<cdiv(262144, 256), 256, 0, stream>>>(encWhh0 + (size_t)d * 262144, eH0pk + (size_t)d * 262144, 256, 256, 262144);
  }
  for (int s = 0; s < 4; ++s) {
    pack_b_kernel<<<cdiv(524288, 256), 256, 0, stream>>>(encWih12 + (size_t)s * 524288, eW12pk + (size_t)s * 524288, 512, 512, 524288);
    pack_b_kernel<<<cdiv(262144, 256), 256, 0, stream>>>(encWhh12 + (size_t)s * 262144, eH12pk + (size_t)s * 262144, 256, 256, 262144);
  }
  pack_b_kernel<<<cdiv(32768, 256), 256, 0, stream>>>(decWih0, dWPpk, 135, 7, 32768);
  for (int s = 0; s < 2; ++s)
    pack_b_kernel<<<cdiv(262144, 256), 256, 0, stream>>>(decWih12 + (size_t)s * 262144, dW12pk + (size_t)s * 262144, 256, 256, 262144);
  for (int s = 0; s < 3; ++s)
    pack_b_kernel<<<cdiv(262144, 256), 256, 0, stream>>>(decWhh + (size_t)s * 262144, dHpk + (size_t)s * 262144, 256, 256, 262144);

  convert_x_kernel<<<cdiv(TSEQ * NB * 32, 256), 256, 0, stream>>>(X, x0pad);
  vadd_kernel<<<cdiv(2048, 256), 256, 0, stream>>>(encBih0, encBhh0, eBias, 2048);
  vadd_kernel<<<cdiv(4096, 256), 256, 0, stream>>>(encBih12, encBhh12, eBias + 2048, 4096);
  vadd_kernel<<<cdiv(3072, 256), 256, 0, stream>>>(decBih, decBhh, dBias, 3072);

  // --- encoder: 3 bidirectional layers ---
  dim3 encGrid(32, 2);
  lstm_enc_kernel<<<encGrid, BLK, 0, stream>>>(x0pad, 32, 32768, eW0pk, eH0pk, eBias, seqA, summ, 0);
  lstm_enc_kernel<<<encGrid, BLK, 0, stream>>>(seqA, 512, 524288, eW12pk, eH12pk, eBias + 2048, seqB, summ, 2);
  lstm_enc_kernel<<<encGrid, BLK, 0, stream>>>(seqB, 512, 524288, eW12pk + (size_t)2 * 524288,
                                               eH12pk + (size_t)2 * 262144, eBias + 4096, seqA, summ, 4);

  // --- VAE bottleneck + heads ---
  fcmu_kernel<<<cdiv(NB * 128, 256), 256, 0, stream>>>(summ, fcMuW, fcMuB, fcLvW, fcLvB, eps, muO, lvO, zbuf);
  mvp_kernel<<<cdiv(NB, 64), 64, 0, stream>>>(zbuf, mvpW1, mvpB1, mvpW2, mvpB2, mvpW3, mvpB3, mxO);
  l2hc_kernel<<<cdiv(NB * 768, 256), 256, 0, stream>>>(zbuf, l2hW, l2hB, l2cW, l2cB, h0s, c0s);
  zih_kernel<<<cdiv(NB * 1024, 256), 256, 0, stream>>>(zbuf, decWih0, zihb);

  // --- autoregressive decoder ---
  lstm_dec_kernel<<<32, BLK, 0, stream>>>(dWPpk, dW12pk, dHpk, dBias, zihb, h0s, c0s, outW, outB, Xhat);
}
